// SAKELayer_73495480369396
// MI455X (gfx1250) — matrix operand order repaired and checked by reading, not compiled, not run on hardware
//
#include <hip/hip_runtime.h>
#include <hip/hip_bf16.h>
#include <math.h>

typedef _Float16 half_t;
typedef __attribute__((ext_vector_type(16))) half_t v16h;
typedef __attribute__((ext_vector_type(8)))  float  v8f;
typedef __attribute__((ext_vector_type(4)))  float  f32x4;

union Frag16 { v16h v; half_t h[16]; f32x4 q[2]; };

__device__ __forceinline__ float celu2f(float x) {
    // celu(x, alpha=2) = x>0 ? x : 2*(exp(x/2)-1)
    return x > 0.f ? x : 2.f * (__expf(0.5f * x) - 1.f);
}

// ---------------------------------------------------------------------------
// Register-resident-B WMMA GEMM:  out = act( A @ W + bias [+ radial*wrad] )
//
// W is pre-reordered into fragment order: tile (kt, ntile) of 32x16 stored as
// 512 contiguous halfs, [n_local(16)][k_local(32)], so a lane's B fragment is
// two contiguous b128 loads. Each wave owns a column strip of NT*16 columns,
// keeps ALL its B fragments (KS*NT) in VGPRs for the entire kernel, and
// grid-strides over PAIRS of 16-row tiles (32 contiguous rows, MT=2).
// Per K-step: one clause of 4 global_load_b128 (A frags for both row tiles),
// then 2*NT back-to-back WMMAs on resident B. No LDS, no barriers.
//
// A fragment layout (16-bit 16x32, ISA 7.12.2):
//  lanes 0-15 : M=lane,    K = {0..7} U {16..23}
//  lanes 16-31: M=lane-16, K = {8..15} U {24..31}
// ---------------------------------------------------------------------------
template <int KS, int NT, bool ACT, bool OUTF16, bool RADIAL>
__global__ __launch_bounds__(256) void k_gemm_breg(
    const half_t* __restrict__ A, int lda,
    const half_t* __restrict__ Wfrag, int ntilesN,
    const float* __restrict__ bias, const float* __restrict__ wrad,
    const float* __restrict__ radial,
    void* __restrict__ outp, int ldo, int col0, int M)
{
    const int lane    = threadIdx.x & 31;
    const int wavesPB = blockDim.x >> 5;
    const int waveId  = blockIdx.x * wavesPB + (threadIdx.x >> 5);
    const int nstrips = ntilesN / NT;
    const int strip   = waveId % nstrips;
    const int rs      = waveId / nstrips;
    const int rowStreams = (gridDim.x * wavesPB) / nstrips;
    const int mtiles  = (M + 15) >> 4;
    const int pairs   = (mtiles + 1) >> 1;

    const int nl   = lane & 15;
    const int mh   = (lane >> 4) << 3;
    const int boff = nl * 32 + ((lane >= 16) ? 8 : 0);
    const int colBase = strip * (NT * 16);

    // ---- load all B fragments for this wave's strip (stay in VGPRs) ----
    Frag16 bfr[KS][NT];
#pragma unroll
    for (int kt = 0; kt < KS; ++kt)
#pragma unroll
        for (int nt = 0; nt < NT; ++nt) {
            const half_t* p = Wfrag +
                (((size_t)(kt * ntilesN + strip * NT + nt)) << 9) + boff;
            bfr[kt][nt].q[0] = *(const f32x4*)(p);
            bfr[kt][nt].q[1] = *(const f32x4*)(p + 16);
        }

    float bv[NT], wr[NT];
#pragma unroll
    for (int nt = 0; nt < NT; ++nt) {
        bv[nt] = bias[colBase + nt * 16 + nl];
        wr[nt] = RADIAL ? wrad[colBase + nt * 16 + nl] : 0.f;
    }

    for (int p = rs; p < pairs; p += rowStreams) {
        const int base = p * 32;
        int r0 = base + nl;       if (r0 >= M) r0 = M - 1;
        int r1 = base + 16 + nl;  if (r1 >= M) r1 = M - 1;
        const int klo = (lane >= 16) ? 8 : 0;
        const half_t* ap0 = A + (size_t)r0 * lda + klo;
        const half_t* ap1 = A + (size_t)r1 * lda + klo;

        // gfx1250 prefetch of next pair's A
        int np = p + rowStreams;
        if (np < pairs) {
            int nr = np * 32 + nl; if (nr >= M) nr = M - 1;
            __builtin_prefetch(A + (size_t)nr * lda, 0, 3);
        }

        v8f acc0[NT] = {}, acc1[NT] = {};
#pragma unroll
        for (int kt = 0; kt < KS; ++kt) {
            Frag16 a0, a1;
            const half_t* p0 = ap0 + kt * 32;
            const half_t* p1 = ap1 + kt * 32;
            a0.q[0] = *(const f32x4*)(p0);
            a0.q[1] = *(const f32x4*)(p0 + 16);
            a1.q[0] = *(const f32x4*)(p1);
            a1.q[1] = *(const f32x4*)(p1 + 16);
#pragma unroll
            for (int nt = 0; nt < NT; ++nt) {
                acc0[nt] = __builtin_amdgcn_wmma_f32_16x16x32_f16(
                    false, a0.v, false, bfr[kt][nt].v, (short)0, acc0[nt],
                    false, false);
                acc1[nt] = __builtin_amdgcn_wmma_f32_16x16x32_f16(
                    false, a1.v, false, bfr[kt][nt].v, (short)0, acc1[nt],
                    false, false);
            }
        }

        // epilogue: C tile: lane n = lane&15, row = j + 8*(lane>=16)
#pragma unroll
        for (int r = 0; r < 2; ++r) {
#pragma unroll
            for (int j = 0; j < 8; ++j) {
                const int m = base + r * 16 + mh + j;
                if (m >= M) continue;
                const float rad = RADIAL ? radial[m] : 0.f;
#pragma unroll
                for (int nt = 0; nt < NT; ++nt) {
                    float v = (r == 0 ? acc0[nt][j] : acc1[nt][j]) + bv[nt];
                    if (RADIAL) v += rad * wr[nt];
                    if (ACT) v = celu2f(v);
                    const size_t oidx =
                        (size_t)m * ldo + col0 + colBase + nt * 16 + nl;
                    if (OUTF16) ((half_t*)outp)[oidx] = (half_t)v;
                    else        ((float*)outp)[oidx]  = v;
                }
            }
        }
    }
}

// ---------------------------------------------------------------------------
// Weight cast + fragment reorder: src fp32 [Ksrc x N] row-major ->
// dst fp16 fragment order; logical row k maps to source row k (+1 if
// k >= skipRow, used to drop the radial row of edge_mlp[0]).
// ---------------------------------------------------------------------------
__global__ void k_cast_w_frag(const float* __restrict__ src, half_t* __restrict__ dst,
                              int K, int N, int skipRow) {
    int t = blockIdx.x * blockDim.x + threadIdx.x;
    if (t >= K * N) return;
    int k = t / N, n = t - k * N;
    int sr = k + (k >= skipRow ? 1 : 0);
    float v = src[(size_t)sr * N + n];
    int kt = k >> 5, kl = k & 31, ntile = n >> 4, nlw = n & 15;
    dst[(((size_t)(kt * (N >> 4) + ntile)) << 9) + nlw * 32 + kl] = (half_t)v;
}

__global__ void k_cast_f16(const float* __restrict__ s, half_t* __restrict__ d, int n) {
    int i = blockIdx.x * blockDim.x + threadIdx.x;
    if (i < n) d[i] = (half_t)s[i];
}

__global__ void k_zero(float* __restrict__ p, int n) {
    int i = blockIdx.x * blockDim.x + threadIdx.x;
    if (i < n) p[i] = 0.f;
}

// Gather Aft[e, 0:256] = [ fp16(h[row[e]]) | fp16(h[col[e]]) ]  (ld = 384)
__global__ void k_gather(const half_t* __restrict__ hf, const int* __restrict__ ei,
                         half_t* __restrict__ Aft, int E) {
    int t = blockIdx.x * blockDim.x + threadIdx.x;
    int e = t >> 5, p = t & 31;
    if (e >= E) return;
    int node = (p < 16) ? ei[e] : ei[E + e];
    int c = (p & 15) * 8;
    f32x4 v = *(const f32x4*)(hf + (size_t)node * 128 + c);
    *(f32x4*)(Aft + (size_t)e * 384 + p * 8) = v;
}

// Aft[e, 256+j] = fp16( (rbf[e] @ Wrbf[:,j] + brbf[j]) * Wf[e,j] )
__global__ void k_rw(const float* __restrict__ rbf, const float* __restrict__ Wr,
                     const float* __restrict__ br, const half_t* __restrict__ Wf,
                     half_t* __restrict__ Aft, int E) {
    int t = blockIdx.x * blockDim.x + threadIdx.x;
    if (t >= E * 128) return;
    int e = t >> 7, j = t & 127;
    float acc = br[j];
    const float* r = rbf + (size_t)e * 18;
#pragma unroll
    for (int k = 0; k < 18; ++k) acc += r[k] * Wr[k * 128 + j];
    float wf = (float)Wf[(size_t)t];
    Aft[(size_t)e * 384 + 256 + j] = (half_t)(acc * wf);
}

// Per-edge: semantic attention, euclid scaling, spatial attention,
// scatter-add into agg (N x 128) and all_aggs (N x 4 x 3). One wave per edge.
__global__ void k_edge_finish(const float* __restrict__ ef_in,
                              const int* __restrict__ ei,
                              const float* __restrict__ radial,
                              const float* __restrict__ cdiff,
                              const float* __restrict__ Wsem,  const float* __restrict__ bsem,
                              const float* __restrict__ Wsem2, const float* __restrict__ bsem2,
                              const float* __restrict__ Wspat, const float* __restrict__ bspat,
                              float* __restrict__ agg, float* __restrict__ all_aggs, int E)
{
    int t = blockIdx.x * blockDim.x + threadIdx.x;
    int e = t >> 5;
    int lane = threadIdx.x & 31;
    if (e >= E) return;

    const float* ef = ef_in + (size_t)e * 128 + lane * 4;
    float e0 = ef[0], e1 = ef[1], e2 = ef[2], e3 = ef[3];
    const int c = lane * 4;

    float s0 = 0.f, s1 = 0.f, s2 = 0.f, s3 = 0.f;
#pragma unroll
    for (int i = 0; i < 4; ++i) {
        f32x4 w = *(const f32x4*)(Wsem + (size_t)(c + i) * 4);
        float x = (i == 0) ? e0 : (i == 1) ? e1 : (i == 2) ? e2 : e3;
        s0 += x * w[0]; s1 += x * w[1]; s2 += x * w[2]; s3 += x * w[3];
    }
#pragma unroll
    for (int m = 16; m >= 1; m >>= 1) {   // wave32 butterfly
        s0 += __shfl_xor(s0, m, 32); s1 += __shfl_xor(s1, m, 32);
        s2 += __shfl_xor(s2, m, 32); s3 += __shfl_xor(s3, m, 32);
    }
    float a0 = celu2f(s0 + bsem[0]);
    float a1 = celu2f(s1 + bsem[1]);
    float a2 = celu2f(s2 + bsem[2]);
    float a3 = celu2f(s3 + bsem[3]);
    float sem = a0 * Wsem2[0] + a1 * Wsem2[1] + a2 * Wsem2[2] + a3 * Wsem2[3] + bsem2[0];

    float rad = radial[e];
    float euclid = 0.5f * (cosf(sqrtf(rad) * 1.57079632679f * 0.5f) + 1.f);
    float scale = sem * euclid;
    e0 *= scale; e1 *= scale; e2 *= scale; e3 *= scale;

    float t0 = 0.f, t1 = 0.f, t2 = 0.f, t3 = 0.f;
#pragma unroll
    for (int i = 0; i < 4; ++i) {
        f32x4 w = *(const f32x4*)(Wspat + (size_t)(c + i) * 4);
        float x = (i == 0) ? e0 : (i == 1) ? e1 : (i == 2) ? e2 : e3;
        t0 += x * w[0]; t1 += x * w[1]; t2 += x * w[2]; t3 += x * w[3];
    }
#pragma unroll
    for (int m = 16; m >= 1; m >>= 1) {
        t0 += __shfl_xor(t0, m, 32); t1 += __shfl_xor(t1, m, 32);
        t2 += __shfl_xor(t2, m, 32); t3 += __shfl_xor(t3, m, 32);
    }
    t0 += bspat[0]; t1 += bspat[1]; t2 += bspat[2]; t3 += bspat[3];

    float x = cdiff[(size_t)e * 3 + 0];
    float y = cdiff[(size_t)e * 3 + 1];
    float z = cdiff[(size_t)e * 3 + 2];
    float nrm = sqrtf(x * x + y * y + z * z);
    float cdx = x / nrm + 1e-8f, cdy = y / nrm + 1e-8f, cdz = z / nrm + 1e-8f;

    const int row = ei[e];
    if (lane < 12) {
        int hh = lane / 3, dd = lane - hh * 3;
        float av = (hh == 0) ? t0 : (hh == 1) ? t1 : (hh == 2) ? t2 : t3;
        float cv = (dd == 0) ? cdx : (dd == 1) ? cdy : cdz;
        atomicAdd(all_aggs + (size_t)row * 12 + lane, av * cv);
    }
    float* ag = agg + (size_t)row * 128 + c;
    atomicAdd(ag + 0, e0); atomicAdd(ag + 1, e1);
    atomicAdd(ag + 2, e2); atomicAdd(ag + 3, e3);
}

// Per node: spat = ||all_aggs||, mu[0] MLP (K=4), build node_in[:,0:256]
__global__ void k_node_pre(const float* __restrict__ all_aggs,
                           const float* __restrict__ agg,
                           const half_t* __restrict__ hf,
                           const float* __restrict__ Wmu0, const float* __restrict__ bmu0,
                           half_t* __restrict__ M0, half_t* __restrict__ node_in, int N)
{
    int t = blockIdx.x * blockDim.x + threadIdx.x;
    if (t >= N * 128) return;
    int n = t >> 7, j = t & 127;
    const float* aa = all_aggs + (size_t)n * 12;
    float sp0 = sqrtf(aa[0]*aa[0] + aa[1]*aa[1] + aa[2]*aa[2]);
    float sp1 = sqrtf(aa[3]*aa[3] + aa[4]*aa[4] + aa[5]*aa[5]);
    float sp2 = sqrtf(aa[6]*aa[6] + aa[7]*aa[7] + aa[8]*aa[8]);
    float sp3 = sqrtf(aa[9]*aa[9] + aa[10]*aa[10] + aa[11]*aa[11]);
    float m0 = bmu0[j] + sp0 * Wmu0[j] + sp1 * Wmu0[128 + j] +
               sp2 * Wmu0[256 + j] + sp3 * Wmu0[384 + j];
    M0[(size_t)n * 128 + j] = (half_t)celu2f(m0);
    node_in[(size_t)n * 384 + j]       = hf[(size_t)n * 128 + j];
    node_in[(size_t)n * 384 + 128 + j] = (half_t)agg[(size_t)n * 128 + j];
}

// ---------------------------------------------------------------------------
// Host-side orchestration
// ---------------------------------------------------------------------------
extern "C" void kernel_launch(void* const* d_in, const int* in_sizes, int n_in,
                              void* d_out, int out_size, void* d_ws, size_t ws_size,
                              hipStream_t stream)
{
    (void)n_in; (void)out_size; (void)ws_size;
    const int N = in_sizes[0] / 128;
    const int E = in_sizes[2];
    const int NOSKIP = 1 << 30;

    const float* h      = (const float*)d_in[0];
    const int*   ei     = (const int*)  d_in[1];
    const float* radial = (const float*)d_in[2];
    const float* cdiff  = (const float*)d_in[3];
    const float* rbf    = (const float*)d_in[4];
    const float* We0   = (const float*)d_in[5];   // [385,128]
    const float* be0   = (const float*)d_in[6];
    const float* We1   = (const float*)d_in[7];   // [128,128]
    const float* be1   = (const float*)d_in[8];
    const float* Wn0   = (const float*)d_in[9];   // [384,128]
    const float* bn0   = (const float*)d_in[10];
    const float* Wn1   = (const float*)d_in[11];  // [128,128]
    const float* bn1   = (const float*)d_in[12];
    const float* Wspat = (const float*)d_in[13];  // [128,4]
    const float* bspat = (const float*)d_in[14];
    const float* Wsem  = (const float*)d_in[15];  // [128,4]
    const float* bsem  = (const float*)d_in[16];
    const float* Wsem2 = (const float*)d_in[17];  // [4,1]
    const float* bsem2 = (const float*)d_in[18];
    const float* Wrbf  = (const float*)d_in[19];  // [18,128]
    const float* brbf  = (const float*)d_in[20];
    const float* Wf1   = (const float*)d_in[21];  // [256,256]
    const float* bf1   = (const float*)d_in[22];
    const float* Wf2   = (const float*)d_in[23];  // [256,128]
    const float* bf2   = (const float*)d_in[24];
    const float* Wmu0  = (const float*)d_in[25];  // [4,128]
    const float* bmu0  = (const float*)d_in[26];
    const float* Wmu1  = (const float*)d_in[27];  // [128,128]
    const float* bmu1  = (const float*)d_in[28];

    // ---- workspace layout ----
    char* ws = (char*)d_ws;
    size_t off = 0;
    auto alloc = [&](size_t bytes) -> char* {
        char* p = ws + off;
        off += (bytes + 255) & ~(size_t)255;
        return p;
    };
    half_t* hf16 = (half_t*)alloc((size_t)N * 128 * 2);
    half_t* buf1 = (half_t*)alloc((size_t)E * 384 * 2);  // Aft(E x 384) -> edge_feat(f32 E x 128)
    half_t* buf2 = (half_t*)alloc((size_t)E * 256 * 2);  // T1(E x 256) -> EF0(E x 128)
    half_t* buf3 = (half_t*)alloc((size_t)E * 128 * 2);  // Wf (filter output)
    float*  zbuf = (float*) alloc((size_t)N * (128 + 12) * 4);
    float*  agg       = zbuf;                   // N x 128
    float*  all_aggs  = zbuf + (size_t)N * 128; // N x 12
    half_t* M0      = (half_t*)alloc((size_t)N * 128 * 2);
    half_t* node_in = (half_t*)alloc((size_t)N * 384 * 2);
    half_t* N0buf   = (half_t*)alloc((size_t)N * 128 * 2);
    half_t* wf1  = (half_t*)alloc(256 * 256 * 2);   // fragment order
    half_t* wf2  = (half_t*)alloc(256 * 128 * 2);
    half_t* we0f = (half_t*)alloc(384 * 128 * 2);   // fused, radial row skipped
    half_t* we1  = (half_t*)alloc(128 * 128 * 2);
    half_t* wn0  = (half_t*)alloc(384 * 128 * 2);
    half_t* wn1  = (half_t*)alloc(128 * 128 * 2);
    half_t* wmu1 = (half_t*)alloc(128 * 128 * 2);

    half_t* Aft       = buf1;                 // E x 384
    float*  edge_feat = (float*)buf1;         // E x 128 f32 (after Aft dead)
    half_t* T1  = buf2;                       // E x 256
    half_t* EF0 = buf2;                       // E x 128 (after T1 dead)
    half_t* Wf  = buf3;                       // E x 128
    const float* wrad = We0 + 256 * 128;      // radial row of edge_mlp[0] weight

    const int TB = 256;
    auto g1 = [&](long long n) { return (unsigned)((n + TB - 1) / TB); };
    const int GE = 1024;  // edge GEMM blocks (8192 waves)
    const int GN = 200;   // node GEMM blocks (1600 waves)

    // ---- 1) casts + fragment reorders ----
    k_cast_f16<<<g1((long long)N * 128), TB, 0, stream>>>(h, hf16, N * 128);
    k_cast_w_frag<<<g1(256 * 256), TB, 0, stream>>>(Wf1, wf1, 256, 256, NOSKIP);
    k_cast_w_frag<<<g1(256 * 128), TB, 0, stream>>>(Wf2, wf2, 256, 128, NOSKIP);
    k_cast_w_frag<<<g1(384 * 128), TB, 0, stream>>>(We0, we0f, 384, 128, 256);
    k_cast_w_frag<<<g1(128 * 128), TB, 0, stream>>>(We1, we1, 128, 128, NOSKIP);
    k_cast_w_frag<<<g1(384 * 128), TB, 0, stream>>>(Wn0, wn0, 384, 128, NOSKIP);
    k_cast_w_frag<<<g1(128 * 128), TB, 0, stream>>>(Wn1, wn1, 128, 128, NOSKIP);
    k_cast_w_frag<<<g1(128 * 128), TB, 0, stream>>>(Wmu1, wmu1, 128, 128, NOSKIP);

    // ---- 2) zero aggregation buffers ----
    k_zero<<<g1((long long)N * 140), TB, 0, stream>>>(zbuf, N * 140);

    // ---- 3) gather edge endpoint features into Aft[:,0:256] ----
    k_gather<<<g1((long long)E * 32), TB, 0, stream>>>(hf16, ei, Aft, E);

    // ---- 4) filter_nn[0]: T1 = celu2(Aft[:,0:256] @ Wf1 + bf1)  [E x 256]
    k_gemm_breg<8, 2, true, true, false><<<GE, TB, 0, stream>>>(
        Aft, 384, wf1, 16, bf1, nullptr, nullptr, T1, 256, 0, E);

    // ---- 5) filter_nn[1]: Wf = T1 @ Wf2 + bf2 (no act)  [E x 128]
    k_gemm_breg<8, 2, false, true, false><<<GE, TB, 0, stream>>>(
        T1, 256, wf2, 8, bf2, nullptr, nullptr, Wf, 128, 0, E);

    // ---- 6) Aft[:,256:384] = fp16(rbf_e * Wf) ----
    k_rw<<<g1((long long)E * 128), TB, 0, stream>>>(rbf, Wrbf, brbf, Wf, Aft, E);

    // ---- 7) edge_mlp[0]: EF0 = celu2(Aft @ we0f + radial*wrad + be0)  K=384
    k_gemm_breg<12, 2, true, true, true><<<GE, TB, 0, stream>>>(
        Aft, 384, we0f, 8, be0, wrad, radial, EF0, 128, 0, E);

    // ---- 8) edge_mlp[1]: edge_feat = celu2(EF0 @ we1 + be1)  [E x 128 f32]
    k_gemm_breg<4, 4, true, false, false><<<GE, TB, 0, stream>>>(
        EF0, 128, we1, 8, be1, nullptr, nullptr, edge_feat, 128, 0, E);

    // ---- 9) attention scaling + segment sums ----
    k_edge_finish<<<g1((long long)E * 32), TB, 0, stream>>>(
        edge_feat, ei, radial, cdiff, Wsem, bsem, Wsem2, bsem2, Wspat, bspat,
        agg, all_aggs, E);

    // ---- 10) per-node: spat norms, mu[0], node_in[:,0:256] ----
    k_node_pre<<<g1((long long)N * 128), TB, 0, stream>>>(
        all_aggs, agg, hf16, Wmu0, bmu0, M0, node_in, N);

    // ---- 11) mu[1]: node_in[:,256:384] = celu2(M0 @ wmu1 + bmu1) ----
    k_gemm_breg<4, 4, true, true, false><<<GN, TB, 0, stream>>>(
        M0, 128, wmu1, 8, bmu1, nullptr, nullptr, node_in, 384, 256, N);

    // ---- 12) node_mlp[0]: N0 = celu2(node_in @ wn0 + bn0)  K=384 ----
    k_gemm_breg<12, 2, true, true, false><<<GN, TB, 0, stream>>>(
        node_in, 384, wn0, 8, bn0, nullptr, nullptr, N0buf, 128, 0, N);

    // ---- 13) node_mlp[1]: out = celu2(N0 @ wn1 + bn1)  [N x 128 f32] ----
    k_gemm_breg<4, 4, true, false, false><<<GN, TB, 0, stream>>>(
        N0buf, 128, wn1, 8, bn1, nullptr, nullptr, d_out, 128, 0, N);
}
